// GDSS_67095979098914
// MI455X (gfx1250) — matrix operand order repaired and checked by reading, not compiled, not run on hardware
//
#include <hip/hip_runtime.h>
#include <hip/hip_bf16.h>
#include <math.h>

// ---------------------------------------------------------------------------
// GDSS loss on MI455X (gfx1250, wave32).
// GEMMs -> v_wmma_f32_16x16x32_f16, 16x64 C-tile per wave (4 WMMA / k-step).
// B operands pre-packed to the per-lane WMMA register layout (one contiguous
// 32B load per fragment). A loaded as contiguous f32 (b128) + cvt_pk. K always
// %32==0 (node embedding zero-padded 16->32). Epilogue flags (accumulate /
// bias / silu) are template parameters -> fully branch-free kernels.
// ---------------------------------------------------------------------------

#define B_ 16
#define N_ 128
#define D_ 16
#define H_ 256
#define L_ 4
#define BMIN_ 0.1f
#define BMAX_ 20.0f

typedef __attribute__((ext_vector_type(16))) _Float16 v16h;
typedef __attribute__((ext_vector_type(8)))  float    v8f;

__device__ __forceinline__ float silu_f(float v) { return v / (1.0f + expf(-v)); }

// ---------------------------------------------------------------------------
// Pack a K x N f32 row-major matrix into the per-lane B layout this GEMM uses (contiguous k per lane half):
//   Bp[(((ktile*ntiles)+ntile)*32 + lane)*16 + t]
//     = B[(ktile*32 + (lane>>4)*16 + t)*N + ntile*16 + (lane&15)]
// Rows k >= Kreal are zero-filled (K padding). Optional batch dim (grid.y).
// ---------------------------------------------------------------------------
__global__ __launch_bounds__(256)
void pack_b(const float* __restrict__ B, _Float16* __restrict__ Bp,
            int K, int Kreal, int N, long sSrc, long sDst) {
  long idx = (long)blockIdx.x * 256 + threadIdx.x;
  if (idx >= (long)K * N) return;
  const float* Bb = B + (long)blockIdx.y * sSrc;
  _Float16* Bpb = Bp + (long)blockIdx.y * sDst;
  int t    = (int)(idx & 15);
  int lane = (int)((idx >> 4) & 31);
  long rest = idx >> 9;
  int ntiles = N >> 4;
  int ntile = (int)(rest % ntiles);
  int ktile = (int)(rest / ntiles);
  int k = ktile * 32 + (lane >> 4) * 16 + t;
  int n = ntile * 16 + (lane & 15);
  Bpb[idx] = (k < Kreal) ? (_Float16)Bb[(long)k * N + n] : (_Float16)0.0f;
}

// ---------------------------------------------------------------------------
// GEMM: C(MxNn) = A(MxK,f32) * Bp(packed f16) [+bias] [+=C] [silu]
// One wave computes a 16 x (16*NT) tile. K % 32 == 0. lda == K.
// ---------------------------------------------------------------------------
template <int NT, int ACC, int BIAS, int SILU>
__global__ __launch_bounds__(32)
void gemm_wmma(const float* __restrict__ A, const _Float16* __restrict__ Bp,
               float* __restrict__ C, const float* __restrict__ bias,
               int K, int Nn, long sA, long sB, long sC) {
  const int lane = threadIdx.x;
  const int hlf  = lane >> 4;
  const int lidx = lane & 15;
  const int tn0  = blockIdx.x * 16 * NT;
  const int tm   = blockIdx.y * 16;
  const int ntiles = Nn >> 4;
  const float*    Ab = A  + (long)blockIdx.z * sA;
  const _Float16* Bb = Bp + (long)blockIdx.z * sB;
  float*          Cb = C  + (long)blockIdx.z * sC;

  v8f c[NT];
#pragma unroll
  for (int q = 0; q < NT; ++q)
#pragma unroll
    for (int i = 0; i < 8; ++i) c[q][i] = 0.0f;

  if (ACC) {
#pragma unroll
    for (int q = 0; q < NT; ++q)
#pragma unroll
      for (int i = 0; i < 8; ++i)
        c[q][i] = Cb[(long)(tm + i + 8 * hlf) * Nn + tn0 + q * 16 + lidx];
  }

  const float* Arow = Ab + (long)(tm + lidx) * K;
  for (int k0 = 0; k0 < K; k0 += 32) {
    v16h a;
    const float* ap = Arow + k0 + hlf * 16;
#pragma unroll
    for (int i = 0; i < 8; ++i) a[i] = (_Float16)ap[i];
#pragma unroll
    for (int i = 0; i < 8; ++i) a[8 + i] = (_Float16)ap[8 + i];
    const _Float16* bk = Bb + ((long)(k0 >> 5) * ntiles) * 512 + lane * 16;
#pragma unroll
    for (int q = 0; q < NT; ++q) {
      const v16h b = *(const v16h*)(bk + (long)(blockIdx.x * NT + q) * 512);
      c[q] = __builtin_amdgcn_wmma_f32_16x16x32_f16(false, a, false, b,
                                                    (short)0, c[q], false, false);
    }
  }

#pragma unroll
  for (int q = 0; q < NT; ++q) {
    const int tn = tn0 + q * 16;
    const float bv = BIAS ? bias[tn + lidx] : 0.0f;
#pragma unroll
    for (int i = 0; i < 8; ++i) {
      float v = c[q][i] + bv;
      if (SILU) v = silu_f(v);
      Cb[(long)(tm + i + 8 * hlf) * Nn + tn + lidx] = v;
    }
  }
}

// --- VPSDE scalars per batch: mc, std, beta; zero loss accumulators ---------
__global__ void scalars_kernel(const float* __restrict__ t,
                               float* __restrict__ sc, float* __restrict__ accum) {
  int b = threadIdx.x;
  if (b < B_) {
    float tb = t[b];
    float ab = expf(-0.5f * (BMIN_ * tb + 0.5f * (BMAX_ - BMIN_) * tb * tb));
    sc[b]          = sqrtf(ab);
    sc[B_ + b]     = sqrtf(1.0f - ab);
    sc[2 * B_ + b] = BMIN_ + tb * (BMAX_ - BMIN_);
  }
  if (threadIdx.x < 2) accum[threadIdx.x] = 0.0f;
}

// --- na (sym, zero diag), a_t, row-normalized a_norm ------------------------
__global__ __launch_bounds__(N_)
void prep_edge(const float* __restrict__ a0, const float* __restrict__ noise_a,
               const float* __restrict__ sc, float* __restrict__ na,
               float* __restrict__ a_t, float* __restrict__ a_norm) {
  int i = blockIdx.x, b = blockIdx.y, j = threadIdx.x;
  long base = ((long)b * N_ + i) * N_;
  float nv = 0.5f * (noise_a[base + j] + noise_a[((long)b * N_ + j) * N_ + i]);
  if (i == j) nv = 0.0f;
  float at = sc[b] * a0[base + j] + sc[B_ + b] * nv;
  na[base + j]  = nv;
  a_t[base + j] = at;
  __shared__ float red[N_];
  red[j] = fabsf(at);
  __syncthreads();
  for (int s = N_ / 2; s > 0; s >>= 1) {
    if (j < s) red[j] += red[j + s];
    __syncthreads();
  }
  float denom = fmaxf(red[0], 1.0f);
  a_norm[base + j] = at / denom;
}

// --- x_t padded to 2048x32 (cols 16..31 zero) -------------------------------
__global__ __launch_bounds__(256)
void prep_x(const float* __restrict__ x0, const float* __restrict__ nx,
            const float* __restrict__ sc, float* __restrict__ x_t) {
  int idx = blockIdx.x * 256 + threadIdx.x;       // over 2048*32
  if (idx >= B_ * N_ * 32) return;
  int col = idx & 31, row = idx >> 5;
  int b = row / N_;
  float v = 0.0f;
  if (col < D_) {
    long s = (long)row * D_ + col;
    v = sc[b] * x0[s] + sc[B_ + b] * nx[s];
  }
  x_t[idx] = v;
}

// --- time MLP: te = silu(t*w1+b1) @ w2 + b2 ---------------------------------
__global__ __launch_bounds__(H_)
void time_embed(const float* __restrict__ t, const float* __restrict__ w1,
                const float* __restrict__ b1, const float* __restrict__ w2,
                const float* __restrict__ b2, float* __restrict__ te) {
  int b = blockIdx.x, h = threadIdx.x;
  __shared__ float sh[H_];
  float v = t[b] * w1[h] + b1[h];
  sh[h] = silu_f(v);
  __syncthreads();
  float acc = b2[h];
  for (int k = 0; k < H_; ++k) acc += sh[k] * w2[k * H_ + h];
  te[b * H_ + h] = acc;
}

// --- fused residual + per-batch vec + bias + LayerNorm (in place) -----------
__global__ __launch_bounds__(H_)
void ln_fuse(float* __restrict__ hbuf, const float* __restrict__ add,
             const float* __restrict__ perb, const float* __restrict__ bias,
             const float* __restrict__ g, const float* __restrict__ beta) {
  int n = blockIdx.x, b = blockIdx.y, h = threadIdx.x;
  long idx = ((long)b * N_ + n) * H_ + h;
  float pre = hbuf[idx] + add[idx] + perb[b * H_ + h] + bias[h];
  __shared__ float red[H_];
  red[h] = pre;
  __syncthreads();
  for (int s = H_ / 2; s > 0; s >>= 1) {
    if (h < s) red[h] += red[h + s];
    __syncthreads();
  }
  float mu = red[0] * (1.0f / H_);
  __syncthreads();
  float d = pre - mu;
  red[h] = d * d;
  __syncthreads();
  for (int s = H_ / 2; s > 0; s >>= 1) {
    if (h < s) red[h] += red[h + s];
    __syncthreads();
  }
  float var = red[0] * (1.0f / H_);
  hbuf[idx] = d * rsqrtf(var + 1e-5f) * g[h] + beta[h];
}

// --- hmean[b,h] = mean_n h_node[b,n,h] --------------------------------------
__global__ __launch_bounds__(H_)
void mean_nodes(const float* __restrict__ h_node, float* __restrict__ hmean) {
  int b = blockIdx.x, h = threadIdx.x;
  float s = 0.0f;
  for (int n = 0; n < N_; ++n) s += h_node[((long)b * N_ + n) * H_ + h];
  hmean[b * H_ + h] = s * (1.0f / N_);
}

// --- loss_x: sum beta*(score_x + noise_x/std)^2 ------------------------------
__global__ __launch_bounds__(256)
void loss_x_kernel(const float* __restrict__ score_x, const float* __restrict__ nx,
                   const float* __restrict__ sc, float* __restrict__ accum) {
  int idx = blockIdx.x * 256 + threadIdx.x;
  int b = idx / (N_ * D_);
  float diff = score_x[idx] + nx[idx] / sc[B_ + b];
  float v = sc[2 * B_ + b] * diff * diff;
  __shared__ float red[256];
  red[threadIdx.x] = v;
  __syncthreads();
  for (int s = 128; s > 0; s >>= 1) {
    if (threadIdx.x < s) red[threadIdx.x] += red[threadIdx.x + s];
    __syncthreads();
  }
  if (threadIdx.x == 0) atomicAdd(&accum[0], red[0]);
}

// --- loss_a over upper-triangle pairs directly from p ------------------------
__global__ __launch_bounds__(H_)
void loss_a_kernel(const float* __restrict__ p, const float* __restrict__ sa_b1,
                   const float* __restrict__ sa_w2, const float* __restrict__ sa_b2,
                   const float* __restrict__ na, const float* __restrict__ sc,
                   float* __restrict__ accum) {
  int i = blockIdx.x, j = blockIdx.y, b = blockIdx.z;
  if (i >= j) return;  // score_a symmetric: 0.5*(p_i+p_j) symmetric in (i,j)
  int h = threadIdx.x;
  float pv = 0.5f * (p[((long)b * N_ + i) * H_ + h] +
                     p[((long)b * N_ + j) * H_ + h]) + sa_b1[h];
  float z = silu_f(pv);
  __shared__ float red[H_];
  red[h] = z * sa_w2[h];
  __syncthreads();
  for (int s = H_ / 2; s > 0; s >>= 1) {
    if (h < s) red[h] += red[h + s];
    __syncthreads();
  }
  if (h == 0) {
    float sa = red[0] + sa_b2[0];
    float target = -na[((long)b * N_ + i) * N_ + j] / sc[B_ + b];
    float d = sa - target;
    atomicAdd(&accum[1], sc[2 * B_ + b] * d * d);
  }
}

__global__ void finalize_kernel(const float* __restrict__ accum, float* __restrict__ out) {
  float lx = accum[0] / (float)(B_ * N_ * D_);
  float la = accum[1] / (float)(B_ * (N_ * (N_ - 1) / 2));
  out[0] = lx + la;
  out[1] = lx;
  out[2] = la;
}

// ---------------------------------------------------------------------------
extern "C" void kernel_launch(void* const* d_in, const int* in_sizes, int n_in,
                              void* d_out, int out_size, void* d_ws, size_t ws_size,
                              hipStream_t stream) {
  (void)in_sizes; (void)n_in; (void)out_size; (void)ws_size;

  // jax pytree flatten order (dict keys alphabetical, layers list in place).
  const float* x_0     = (const float*)d_in[0];
  const float* a_0     = (const float*)d_in[1];
  const float* t_in    = (const float*)d_in[2];
  const float* noise_x = (const float*)d_in[3];
  const float* noise_a = (const float*)d_in[4];
  const float* adj_b   = (const float*)d_in[5];
  const float* adj_w   = (const float*)d_in[6];
  const float* node_b  = (const float*)d_in[47];
  const float* node_w  = (const float*)d_in[48];
  const float* sa_b1   = (const float*)d_in[49];
  const float* sa_b2   = (const float*)d_in[50];
  const float* sa_w1   = (const float*)d_in[51];
  const float* sa_w2   = (const float*)d_in[52];
  const float* sx_b1   = (const float*)d_in[53];
  const float* sx_b2   = (const float*)d_in[54];
  const float* sx_w1   = (const float*)d_in[55];
  const float* sx_w2   = (const float*)d_in[56];
  const float* time_b1 = (const float*)d_in[57];
  const float* time_b2 = (const float*)d_in[58];
  const float* time_w1 = (const float*)d_in[59];
  const float* time_w2 = (const float*)d_in[60];

  const size_t SZ_BND = (size_t)B_ * N_ * D_;   // 32768
  const size_t SZ_BNN = (size_t)B_ * N_ * N_;   // 262144
  const size_t SZ_BNH = (size_t)B_ * N_ * H_;   // 524288
  const size_t SZ_BH  = (size_t)B_ * H_;        // 4096

  float* ws = (float*)d_ws;
  float* sc       = ws;                      // 64
  float* accum    = ws + 64;                 // 16
  float* te       = ws + 128;
  float* teproj   = te + SZ_BH;
  float* meanproj = teproj + SZ_BH;
  float* hmean    = meanproj + SZ_BH;
  float* x_t      = hmean + SZ_BH;           // padded 2048x32
  float* score_x  = x_t + (size_t)B_ * N_ * 32;
  float* na       = score_x + SZ_BND;
  float* a_t      = na + SZ_BNN;
  float* a_norm   = a_t + SZ_BNN;
  float* h_node   = a_norm + SZ_BNN;
  float* h_edge   = h_node + SZ_BNH;
  float* tmp      = h_edge + SZ_BNH;
  float* tmp2     = tmp + SZ_BNH;
  float* msg      = tmp2 + SZ_BNH;
  float* p        = msg + SZ_BNH;
  // packed f16 regions (2 halfs per float slot)
  _Float16* pW3   = (_Float16*)(p + SZ_BNH);           // 768x256 = 196608 h
  _Float16* pW1   = pW3 + (size_t)768 * 256;           // 256x256 = 65536 h
  _Float16* pMisc = pW1 + (size_t)256 * 256;           // up to 128x256 = 32768 h
  _Float16* pTmp  = pMisc + (size_t)128 * 256;         // B x (128x256) = 524288 h

  auto pack = [&](const float* Bsrc, _Float16* Bdst, int K, int Kreal, int Nn,
                  int batch, long sSrc, long sDst) {
    long total = (long)K * Nn;
    dim3 grid((unsigned)((total + 255) / 256), batch);
    hipLaunchKernelGGL(pack_b, grid, dim3(256), 0, stream,
                       Bsrc, Bdst, K, Kreal, Nn, sSrc, sDst);
  };
  // NT=4, plain (no bias / no acc / no silu)
  auto gemm4 = [&](const float* A, const _Float16* Bp, float* C,
                   int M, int K, int batch, long sA, long sB, long sC) {
    dim3 grid(H_ / 64, M / 16, batch);
    hipLaunchKernelGGL((gemm_wmma<4, 0, 0, 0>), grid, dim3(32), 0, stream,
                       A, Bp, C, (const float*)nullptr, K, H_, sA, sB, sC);
  };
  // NT=4, += accumulate into C
  auto gemm4_acc = [&](const float* A, const _Float16* Bp, float* C, int M, int K) {
    dim3 grid(H_ / 64, M / 16, 1);
    hipLaunchKernelGGL((gemm_wmma<4, 1, 0, 0>), grid, dim3(32), 0, stream,
                       A, Bp, C, (const float*)nullptr, K, H_, 0L, 0L, 0L);
  };
  // NT=4, +bias
  auto gemm4_b = [&](const float* A, const _Float16* Bp, float* C,
                     const float* bias, int M, int K) {
    dim3 grid(H_ / 64, M / 16, 1);
    hipLaunchKernelGGL((gemm_wmma<4, 0, 1, 0>), grid, dim3(32), 0, stream,
                       A, Bp, C, bias, K, H_, 0L, 0L, 0L);
  };
  // NT=4, +bias +silu
  auto gemm4_bs = [&](const float* A, const _Float16* Bp, float* C,
                      const float* bias, int M, int K) {
    dim3 grid(H_ / 64, M / 16, 1);
    hipLaunchKernelGGL((gemm_wmma<4, 0, 1, 1>), grid, dim3(32), 0, stream,
                       A, Bp, C, bias, K, H_, 0L, 0L, 0L);
  };
  // NT=1, +bias (narrow output, Nn=16)
  auto gemm1_b = [&](const float* A, const _Float16* Bp, float* C,
                     const float* bias, int M, int K, int Nn) {
    dim3 grid(Nn / 16, M / 16, 1);
    hipLaunchKernelGGL((gemm_wmma<1, 0, 1, 0>), grid, dim3(32), 0, stream,
                       A, Bp, C, bias, K, Nn, 0L, 0L, 0L);
  };

  // 1) scalars + accumulator init
  hipLaunchKernelGGL(scalars_kernel, dim3(1), dim3(32), 0, stream, t_in, sc, accum);
  // 2) edge prep
  hipLaunchKernelGGL(prep_edge, dim3(N_, B_), dim3(N_), 0, stream,
                     a_0, noise_a, sc, na, a_t, a_norm);
  // 3) x_t (padded to K=32)
  hipLaunchKernelGGL(prep_x, dim3((B_ * N_ * 32) / 256), dim3(256), 0, stream,
                     x_0, noise_x, sc, x_t);
  // 4) time embedding
  hipLaunchKernelGGL(time_embed, dim3(B_), dim3(H_), 0, stream,
                     t_in, time_w1, time_b1, time_w2, time_b2, te);
  // 5) h_node = x_t @ node_w + node_b   (K padded 16->32)
  pack(node_w, pMisc, 32, D_, H_, 1, 0, 0);
  gemm4_b(x_t, pMisc, h_node, node_b, B_ * N_, 32);
  // 6) h_edge = a_t @ adj_w + adj_b
  pack(adj_w, pMisc, N_, N_, H_, 1, 0, 0);
  gemm4_b(a_t, pMisc, h_edge, adj_b, B_ * N_, N_);

  // 7) GNN layers
  for (int l = 0; l < L_; ++l) {
    const float* msg_b = (const float*)d_in[7 + 10 * l + 0];
    const float* msg_w = (const float*)d_in[7 + 10 * l + 1];
    const float* ne_b  = (const float*)d_in[7 + 10 * l + 2];
    const float* ne_g  = (const float*)d_in[7 + 10 * l + 3];
    const float* nn_b  = (const float*)d_in[7 + 10 * l + 4];
    const float* nn_g  = (const float*)d_in[7 + 10 * l + 5];
    const float* ue_b  = (const float*)d_in[7 + 10 * l + 6];
    const float* ue_w  = (const float*)d_in[7 + 10 * l + 7];
    const float* un_b  = (const float*)d_in[7 + 10 * l + 8];
    const float* un_w  = (const float*)d_in[7 + 10 * l + 9];

    // tmp = h_node @ msg_w + msg_b
    pack(msg_w, pW1, H_, H_, H_, 1, 0, 0);
    gemm4_b(h_node, pW1, tmp, msg_b, B_ * N_, H_);
    // msg[b] = a_norm[b] @ tmp[b]   (pack tmp per batch as B operand)
    pack(tmp, pTmp, N_, N_, H_, B_, (long)N_ * H_, (long)N_ * H_);
    gemm4(a_norm, pTmp, msg, N_, N_, B_,
          (long)N_ * N_, (long)N_ * H_, (long)N_ * H_);
    // un_w packed once; sub-blocks are ktile offsets (256 rows = 8 ktiles)
    pack(un_w, pW3, 3 * H_, 3 * H_, H_, 1, 0, 0);
    const long sub = (long)8 * (H_ / 16) * 512;   // 65536 halfs per 256 rows
    gemm4(h_node, pW3, tmp2, B_ * N_, H_, 1, 0L, 0L, 0L);
    gemm4_acc(msg, pW3 + sub, tmp2, B_ * N_, H_);
    gemm4(te, pW3 + 2 * sub, teproj, B_, H_, 1, 0L, 0L, 0L);
    hipLaunchKernelGGL(ln_fuse, dim3(N_, B_), dim3(H_), 0, stream,
                       h_node, tmp2, teproj, un_b, nn_g, nn_b);
    // edge update
    hipLaunchKernelGGL(mean_nodes, dim3(B_), dim3(H_), 0, stream, h_node, hmean);
    pack(ue_w, pW3, 3 * H_, 3 * H_, H_, 1, 0, 0);
    gemm4(h_node, pW3, tmp2, B_ * N_, H_, 1, 0L, 0L, 0L);
    gemm4_acc(h_edge, pW3 + 2 * sub, tmp2, B_ * N_, H_);
    gemm4(hmean, pW3 + sub, meanproj, B_, H_, 1, 0L, 0L, 0L);
    hipLaunchKernelGGL(ln_fuse, dim3(N_, B_), dim3(H_), 0, stream,
                       h_edge, tmp2, meanproj, ue_b, ne_g, ne_b);
  }

  // 8) score_x = silu(h_node @ sx_w1 + sx_b1) @ sx_w2 + sx_b2
  pack(sx_w1, pW1, H_, H_, H_, 1, 0, 0);
  gemm4_bs(h_node, pW1, tmp, sx_b1, B_ * N_, H_);
  pack(sx_w2, pMisc, H_, H_, D_, 1, 0, 0);
  gemm1_b(tmp, pMisc, score_x, sx_b2, B_ * N_, H_, D_);
  // 9) p = h_node @ sa_w1
  pack(sa_w1, pW1, H_, H_, H_, 1, 0, 0);
  gemm4(h_node, pW1, p, B_ * N_, H_, 1, 0L, 0L, 0L);

  // 10) losses
  hipLaunchKernelGGL(loss_x_kernel, dim3((B_ * N_ * D_) / 256), dim3(256), 0, stream,
                     score_x, noise_x, sc, accum);
  hipLaunchKernelGGL(loss_a_kernel, dim3(N_, N_, B_), dim3(H_), 0, stream,
                     p, sa_b1, sa_w2, sa_b2, na, sc, accum);
  // 11) finalize
  hipLaunchKernelGGL(finalize_kernel, dim3(1), dim3(1), 0, stream,
                     accum, (float*)d_out);
}
